// A3C_PerCellModel_4Neighbor_51866025066822
// MI455X (gfx1250) — matrix operand order repaired and checked by reading, not compiled, not run on hardware
//
#include <hip/hip_runtime.h>

typedef _Float16 f16;
typedef __attribute__((ext_vector_type(16))) _Float16 v16h;
typedef __attribute__((ext_vector_type(8)))  float    v8f;
typedef __attribute__((ext_vector_type(4)))  unsigned v4u;

#define HH 512
#define WW 512
#define HWPIX (HH*WW)        // 2^18
#define PW 514               // padded width
#define PLANE (514*514)      // padded plane elements
#define CIN 14
#define NCELLS 8192
#define BPG 128              // stat blocks per group
#define RSH 40               // LDS B-tile row stride in halves (80 B; 16B aligned, bank-spread)

// ---------------- zero fill ----------------
__global__ void zero_k(float* p, long n) {
    long i = (long)blockIdx.x * blockDim.x + threadIdx.x;
    if (i < n) p[i] = 0.f;
}

// ---------------- zero the 1-pixel halo of C padded planes ----------------
__global__ void hzero_k(f16* a, int C) {
    int t = blockIdx.x * blockDim.x + threadIdx.x;
    int per = 4 * PW;
    if (t >= C * per) return;
    int c = t / per, r = t % per;
    int seg = r / PW, o = r % PW;
    long base = (long)c * PLANE;
    long addr;
    if (seg == 0)      addr = base + o;                       // row 0
    else if (seg == 1) addr = base + (long)(PW - 1) * PW + o; // row 513
    else if (seg == 2) addr = base + (long)o * PW;            // col 0
    else               addr = base + (long)o * PW + (PW - 1); // col 513
    a[addr] = (f16)0.f;
}

// ---------------- pad input x -> f16 padded planes (halo zero) ----------------
__global__ void padx_k(const float* __restrict__ x, f16* __restrict__ xp) {
    long t = (long)blockIdx.x * blockDim.x + threadIdx.x;
    if (t >= (long)CIN * PLANE) return;
    int c = (int)(t / PLANE);
    int rem = (int)(t % PLANE);
    int row = rem / PW, col = rem % PW;
    f16 v = (f16)0.f;
    if (row >= 1 && row <= HH && col >= 1 && col <= WW)
        v = (f16)x[(long)c * HWPIX + (row - 1) * WW + (col - 1)];
    xp[t] = v;
}

// ------------- pack weights into WMMA A-fragment layout -------------
// A[m][k] = src[m*strideM + k*strideK]; zero-filled past K so B over-reads are harmless.
__global__ void pack_a_k(const float* __restrict__ src, unsigned* __restrict__ dst,
                         int M, int K, int strideM, int strideK, int nsteps) {
    int idx = blockIdx.x * blockDim.x + threadIdx.x;
    int total = (M >> 4) * nsteps * 256;
    if (idx >= total) return;
    int j = idx & 7;
    int lane = (idx >> 3) & 31;
    int rest = idx >> 8;
    int s = rest % nsteps;
    int t = rest / nsteps;
    int m = t * 16 + (lane & 15);
    int kb = s * 32 + ((lane >> 4) * 16) + j * 2;
    union { f16 h[2]; unsigned u; } pk;
    #pragma unroll
    for (int e = 0; e < 2; e++) {
        int k = kb + e;
        float v = (k < K) ? src[(long)m * strideM + (long)k * strideK] : 0.f;
        pk.h[e] = (f16)v;
    }
    dst[idx] = pk.u;
}

// ---------------- conv1: direct (C_IN=14), branchless on padded x ----------------
__global__ void conv1_k(const f16* __restrict__ xp, const float* __restrict__ w1,
                        const float* __restrict__ b1, f16* __restrict__ yp) {
    __shared__ float wsm[CIN * 9];
    __shared__ float bsh;
    int oc = blockIdx.y;
    for (int i = threadIdx.x; i < CIN * 9; i += blockDim.x) wsm[i] = w1[oc * CIN * 9 + i];
    if (threadIdx.x == 0) bsh = b1[oc];
    __syncthreads();
    int pix = blockIdx.x * blockDim.x + threadIdx.x;
    int py = pix >> 9, px = pix & 511;
    int base = py * PW + px;   // padded (py+dy, px+dx) = base + dy*PW + dx
    float acc = bsh;
    for (int ic = 0; ic < CIN; ic++) {
        int pbase = ic * PLANE + base;
        #pragma unroll
        for (int dy = 0; dy < 3; dy++)
            #pragma unroll
            for (int dx = 0; dx < 3; dx++)
                acc += wsm[ic * 9 + dy * 3 + dx] * (float)xp[pbase + dy * PW + dx];
    }
    yp[(long)oc * PLANE + (py + 1) * PW + (px + 1)] = (f16)acc;
}

// -------------- group stats over padded interior: per-(group,block) partials --------------
__global__ void gstats_k(const f16* __restrict__ a, float* __restrict__ part, int cpg) {
    int g = blockIdx.x / BPG, b = blockIdx.x % BPG;
    long glen = (long)cpg * HWPIX;
    float s = 0.f, sq = 0.f;
    for (long i = (long)b * blockDim.x + threadIdx.x; i < glen; i += (long)BPG * blockDim.x) {
        int c = g * cpg + (int)(i >> 18);
        int pix = (int)(i & (HWPIX - 1));
        float v = (float)a[(long)c * PLANE + ((pix >> 9) + 1) * PW + (pix & 511) + 1];
        s += v; sq += v * v;
    }
    __shared__ float rs[256], rq[256];
    rs[threadIdx.x] = s; rq[threadIdx.x] = sq;
    __syncthreads();
    for (int o = 128; o > 0; o >>= 1) {
        if ((int)threadIdx.x < o) { rs[threadIdx.x] += rs[threadIdx.x + o]; rq[threadIdx.x] += rq[threadIdx.x + o]; }
        __syncthreads();
    }
    if (threadIdx.x == 0) { part[(g * BPG + b) * 2] = rs[0]; part[(g * BPG + b) * 2 + 1] = rq[0]; }
}

__global__ void gfinal_k(const float* __restrict__ part, float* __restrict__ stats, int G) {
    int g = blockIdx.x * blockDim.x + threadIdx.x;
    if (g >= G) return;
    float s = 0.f, sq = 0.f;
    for (int b = 0; b < BPG; b++) { s += part[(g * BPG + b) * 2]; sq += part[(g * BPG + b) * 2 + 1]; }
    stats[2 * g] = s; stats[2 * g + 1] = sq;
}

// -------------- GroupNorm apply + ReLU, in place on padded buffer --------------
__global__ void gn_apply_k(f16* __restrict__ a, const float* __restrict__ stats,
                           const float* __restrict__ gamma, const float* __restrict__ beta,
                           int cpg) {
    long idx = (long)blockIdx.x * blockDim.x + threadIdx.x;
    int c = (int)(idx >> 18);
    int g = c / cpg;
    int pix = (int)(idx & (HWPIX - 1));
    long addr = (long)c * PLANE + ((pix >> 9) + 1) * PW + (pix & 511) + 1;
    float cnt = (float)cpg * (float)HWPIX;
    float mean = stats[2 * g] / cnt;
    float var = stats[2 * g + 1] / cnt - mean * mean;
    float sc = rsqrtf(var + 1e-5f) * gamma[c];
    float v = ((float)a[addr] - mean) * sc + beta[c];
    a[addr] = (f16)fmaxf(v, 0.f);
}

// -------------- WMMA GEMM with cooperative LDS B staging --------------
// Block = (M/16) waves sharing one 64-wide N strip. Per K-step:
//   stage 32Kx64N B tile into LDS (branchless, padded source, double-buffered),
//   each wave: 2x global b128 (packed A, prefetched a step ahead) + 8x ds_load_b128 + 4x WMMA.
template <int MODE>
__global__ void gemm_k(const v4u* __restrict__ packA, const f16* __restrict__ Bsrc,
                       const float* __restrict__ bias, f16* __restrict__ out,
                       int M, int nsteps,
                       const int* __restrict__ cell_i, const int* __restrict__ cell_j,
                       int Kld, int relu) {
    __shared__ f16 sb[2][64 * RSH];
    int tid = threadIdx.x;
    int wave = tid >> 5, lane = tid & 31;
    int mtile = wave;
    long nbase = (long)blockIdx.x * 64;
    int ln = lane & 15;
    int kh = (lane >> 4) << 4;   // B K-offset within step: 0 or 16

    v8f acc[4] = {};

    // per-thread staging base (n fixed per thread: stride is a multiple of 64)
    int base_n = 0;
    if (MODE == 0) {
        int n = (int)nbase + (tid & 63);
        base_n = (n >> 9) * PW + (n & 511);
    } else if (MODE == 1) {
        int n = (int)nbase + (tid & 63);
        base_n = cell_i[n] * PW + cell_j[n];
    }

    int ppy[4], ppx[4];
    if (MODE == 0) {
        #pragma unroll
        for (int sub = 0; sub < 4; sub++) {
            int n = (int)nbase + sub * 16 + ln;
            ppy[sub] = n >> 9; ppx[sub] = n & 511;
        }
    }

    if (MODE != 2) {
        for (int e = tid; e < 2048; e += blockDim.x) {
            int kl = e >> 6;
            int c = kl / 9, r = kl - 9 * c;
            sb[0][(e & 63) * RSH + kl] = Bsrc[c * PLANE + base_n + (r / 3) * PW + (r - (r / 3) * 3)];
        }
        __syncthreads();
    }

    for (int s = 0; s < nsteps; s++) {
        int pp = s & 1;
        if (MODE != 2 && s + 1 < nsteps) {
            int ks = (s + 1) * 32;
            for (int e = tid; e < 2048; e += blockDim.x) {
                int gk = ks + (e >> 6);
                int c = gk / 9, r = gk - 9 * c;
                sb[pp ^ 1][(e & 63) * RSH + (e >> 6)] =
                    Bsrc[c * PLANE + base_n + (r / 3) * PW + (r - (r / 3) * 3)];
            }
        }

        union { v4u q[2]; v16h h; } A;
        const v4u* ap = packA + ((long)(mtile * nsteps + s) * 32 + lane) * 2;
        A.q[0] = ap[0]; A.q[1] = ap[1];
        if (s + 1 < nsteps) {
            // warm next step's packed-A tile (global_prefetch_b8)
            __builtin_prefetch((const void*)(packA + ((long)(mtile * nsteps + s + 1) * 32 + lane) * 2), 0, 3);
        }

        #pragma unroll
        for (int sub = 0; sub < 4; sub++) {
            union { v4u q[2]; v16h h; } Bu;
            if (MODE == 2) {
                long n = nbase + sub * 16 + ln;
                const v4u* rp = (const v4u*)(Bsrc + n * Kld + s * 32 + kh);
                Bu.q[0] = rp[0]; Bu.q[1] = rp[1];
            } else {
                const v4u* bp = (const v4u*)(&sb[pp][(sub * 16 + ln) * RSH + kh]);
                Bu.q[0] = bp[0]; Bu.q[1] = bp[1];
            }
            acc[sub] = __builtin_amdgcn_wmma_f32_16x16x32_f16(
                false, A.h, false, Bu.h, (short)0, acc[sub], false, false);
        }
        if (MODE != 2) __syncthreads();
    }

    int mrow0 = mtile * 16 + ((lane >> 4) << 3);
    #pragma unroll
    for (int sub = 0; sub < 4; sub++) {
        long n = nbase + sub * 16 + ln;
        #pragma unroll
        for (int r = 0; r < 8; r++) {
            int m = mrow0 + r;
            float v = acc[sub][r] + bias[m];
            if (relu) v = fmaxf(v, 0.f);
            if (MODE == 0)
                out[(long)m * PLANE + (ppy[sub] + 1) * PW + (ppx[sub] + 1)] = (f16)v;
            else
                out[n * M + m] = (f16)v;
        }
    }
}

// -------------- value head: pooled -> 64 -> 1 --------------
__global__ void value_k(const float* __restrict__ chstats,
                        const float* __restrict__ vw1, const float* __restrict__ vb1,
                        const float* __restrict__ vw2, const float* __restrict__ vb2,
                        float* __restrict__ outv) {
    __shared__ float pooled[128];
    __shared__ float hr[64];
    int t = threadIdx.x;   // 64 threads
    pooled[t]      = chstats[2 * t]        / (float)HWPIX;
    pooled[t + 64] = chstats[2 * (t + 64)] / (float)HWPIX;
    __syncthreads();
    float h = vb1[t];
    for (int c = 0; c < 128; c++) h += pooled[c] * vw1[c * 64 + t];
    hr[t] = fmaxf(h, 0.f) * vw2[t];
    __syncthreads();
    if (t == 0) {
        float s = vb2[0];
        for (int j = 0; j < 64; j++) s += hr[j];
        outv[0] = s;
    }
}

// -------------- TDM (Tensor Data Mover) 1-D global->LDS copy --------------
#if __has_builtin(__builtin_amdgcn_tensor_load_to_lds) && __has_builtin(__builtin_amdgcn_s_wait_tensorcnt)
#define HAVE_TDM 1
typedef __attribute__((ext_vector_type(4))) unsigned tdm_u4;
typedef __attribute__((ext_vector_type(8))) int      tdm_i8;
typedef __attribute__((ext_vector_type(4))) int      tdm_i4;

// D# per CDNA5 ISA §8.3/§8.4: group0 {count=1, lds_addr, global_addr[56:0], type=2},
// group1 {data_size=4B, tensor_dim0=tile_dim0=nelem, 1-D tile}.
__device__ __forceinline__ void tdm_copy_f32(const float* src, void* lds_dst, unsigned nelem) {
    unsigned lds = (unsigned)(size_t)lds_dst;                 // LDS aperture: low 32 bits = LDS offset
    unsigned long long ga = (unsigned long long)(size_t)src;
    tdm_u4 g0;
    g0[0] = 1u;                                               // count=1
    g0[1] = lds;                                              // lds_addr (bytes)
    g0[2] = (unsigned)(ga & 0xFFFFFFFFull);                   // global_addr[31:0]  (bits 95:64)
    g0[3] = (unsigned)((ga >> 32) & 0x01FFFFFFull) | (2u << 30); // global_addr[56:32] | type=2
    tdm_i8 g1;
    g1[0] = (int)(2u << 16);                                  // wg_mask=0, data_size=2 (4 bytes)
    g1[1] = (int)(nelem << 16);                               // tensor_dim0 low16 (bits 63:48)
    g1[2] = (int)((nelem >> 16) | (1u << 16));                // tensor_dim0 hi16 | tensor_dim1=1
    g1[3] = (int)(nelem << 16);                               // tensor_dim1 hi(0) | tile_dim0
    g1[4] = 0;                                                // tile_dim1=0, tile_dim2=0 (unused)
    g1[5] = (int)nelem;                                       // tensor_dim0_stride low32
    g1[6] = 0; g1[7] = 0;
    tdm_i4 z4 = {0, 0, 0, 0};
#if defined(__clang_major__) && __clang_major__ >= 23
    tdm_i8 z8 = {0, 0, 0, 0, 0, 0, 0, 0};
    __builtin_amdgcn_tensor_load_to_lds(g0, g1, z4, z4, z8, 0);
#else
    __builtin_amdgcn_tensor_load_to_lds(g0, g1, z4, z4, 0);
#endif
}
#endif

// -------------- heads + deterministic logprob/entropy partials + scatter --------------
__global__ void heads_k(const f16* __restrict__ shrow, const int* __restrict__ cell_i,
                        const int* __restrict__ cell_j, const int* __restrict__ action,
                        const float* __restrict__ bw, const float* __restrict__ bbv,
                        const float* __restrict__ iw, const float* __restrict__ ibv,
                        const float* __restrict__ tw, const float* __restrict__ tbv,
                        float* __restrict__ out, float* __restrict__ hpart) {
    __shared__ float WB[512], WI[512], WT[512];
    __shared__ float red[256];
#ifdef HAVE_TDM
    if (threadIdx.x < 32) {                 // one wave issues TDM (EXEC ignored by TDM)
        if (threadIdx.x == 0) {
            tdm_copy_f32(bw, WB, 512);
            tdm_copy_f32(iw, WI, 512);
            tdm_copy_f32(tw, WT, 512);
        }
        __builtin_amdgcn_s_wait_tensorcnt(0);
    }
    __syncthreads();
#else
    for (int i = threadIdx.x; i < 512; i += blockDim.x) { WB[i] = bw[i]; WI[i] = iw[i]; WT[i] = tw[i]; }
    __syncthreads();
#endif
    int cell = blockIdx.x * blockDim.x + threadIdx.x;
    float zb[4], zi[4], zt[4];
    #pragma unroll
    for (int k = 0; k < 4; k++) { zb[k] = bbv[k]; zi[k] = ibv[k]; zt[k] = tbv[k]; }
    const f16* row = shrow + (long)cell * 128;
    for (int c = 0; c < 128; c++) {
        float v = (float)row[c];
        #pragma unroll
        for (int k = 0; k < 4; k++) {
            zb[k] += v * WB[c * 4 + k];
            zi[k] += v * WI[c * 4 + k];
            zt[k] += v * WT[c * 4 + k];
        }
    }
    float lp = 0.f, en = 0.f;
    int ci = cell_i[cell], cj = cell_j[cell];
    const int di[4] = {-1, 0, 1, 0};
    const int dj[4] = {0, 1, 0, -1};
    #pragma unroll
    for (int k = 0; k < 4; k++) {
        float p = 1.f / (1.f + expf(-zb[k]));
        p = fminf(fmaxf(p, 1e-7f), 1.f - 1e-7f);
        float a = (float)action[cell * 4 + k];
        float lpp = logf(p), lq = logf(1.f - p);
        lp += a * lpp + (1.f - a) * lq;
        en -= p * lpp + (1.f - p) * lq;
        out[262147 + cell * 4 + k] = zi[k];
        out[294915 + cell * 4 + k] = zt[k];
        if (a > 0.5f) {
            int ni = ci + di[k], nj = cj + dj[k];
            if (ni >= 0 && ni < HH && nj >= 0 && nj < WW) out[ni * WW + nj] = 1.0f;
        }
    }
    red[threadIdx.x] = lp; __syncthreads();
    for (int o = 128; o > 0; o >>= 1) { if ((int)threadIdx.x < o) red[threadIdx.x] += red[threadIdx.x + o]; __syncthreads(); }
    if (threadIdx.x == 0) hpart[blockIdx.x * 2] = red[0];
    __syncthreads();
    red[threadIdx.x] = en; __syncthreads();
    for (int o = 128; o > 0; o >>= 1) { if ((int)threadIdx.x < o) red[threadIdx.x] += red[threadIdx.x + o]; __syncthreads(); }
    if (threadIdx.x == 0) hpart[blockIdx.x * 2 + 1] = red[0];
}

__global__ void lpent_final_k(const float* __restrict__ hpart, float* __restrict__ out, int nb) {
    if (threadIdx.x == 0) {
        float lp = 0.f, en = 0.f;
        for (int b = 0; b < nb; b++) { lp += hpart[2 * b]; en += hpart[2 * b + 1]; }
        out[262144] = lp;
        out[262145] = en;
    }
}

// ---------------------------------------------------------------------------
extern "C" void kernel_launch(void* const* d_in, const int* in_sizes, int n_in,
                              void* d_out, int out_size, void* d_ws, size_t ws_size,
                              hipStream_t stream) {
    (void)in_sizes; (void)n_in; (void)out_size; (void)ws_size;
    const float* x      = (const float*)d_in[0];
    const int* cell_i   = (const int*)d_in[1];
    const int* cell_j   = (const int*)d_in[2];
    const int* action   = (const int*)d_in[3];
    const float* w1 = (const float*)d_in[4];  const float* b1 = (const float*)d_in[5];
    const float* g1 = (const float*)d_in[6];  const float* be1 = (const float*)d_in[7];
    const float* w2 = (const float*)d_in[8];  const float* b2 = (const float*)d_in[9];
    const float* g2 = (const float*)d_in[10]; const float* be2 = (const float*)d_in[11];
    const float* w3 = (const float*)d_in[12]; const float* b3 = (const float*)d_in[13];
    const float* g3 = (const float*)d_in[14]; const float* be3 = (const float*)d_in[15];
    const float* fw1 = (const float*)d_in[16]; const float* fb1 = (const float*)d_in[17];
    const float* fw2 = (const float*)d_in[18]; const float* fb2 = (const float*)d_in[19];
    const float* bw = (const float*)d_in[20]; const float* bb = (const float*)d_in[21];
    const float* iw = (const float*)d_in[22]; const float* ib = (const float*)d_in[23];
    const float* tw = (const float*)d_in[24]; const float* tb = (const float*)d_in[25];
    const float* vw1 = (const float*)d_in[26]; const float* vb1 = (const float*)d_in[27];
    const float* vw2 = (const float*)d_in[28]; const float* vb2 = (const float*)d_in[29];
    float* out = (float*)d_out;
    char* ws = (char*)d_ws;

    const size_t MB = 1ull << 20;
    // R0 (68 MiB): ap1 (48+2 slack planes), later ap3 (128 planes)
    // R1 (52 MiB): ap2 (96 planes)
    f16* ap1   = (f16*)(ws + 0 * MB);
    f16* ap3   = (f16*)(ws + 0 * MB);
    f16* ap2   = (f16*)(ws + 68 * MB);
    f16* xpad  = (f16*)(ws + 120 * MB);
    unsigned* pw2 = (unsigned*)(ws + 128 * MB);
    unsigned* pw3 = (unsigned*)(ws + 129 * MB);
    unsigned* pf1 = (unsigned*)(ws + 130 * MB);
    unsigned* pf2 = (unsigned*)(ws + 131 * MB);
    float* stats  = (float*)(ws + 132 * MB);
    float* part   = (float*)(ws + 132 * MB + 4096);
    float* hpart  = (float*)(ws + 132 * MB + 4096 + 131072);
    f16* hid1     = (f16*)(ws + 133 * MB);     // 8192*256 f16
    f16* shbuf    = (f16*)(ws + 137 * MB);     // 8192*128 f16

    zero_k<<<(HWPIX + 255) / 256, 256, 0, stream>>>(out, HWPIX);
    hzero_k<<<(128 * 4 * PW + 255) / 256, 256, 0, stream>>>((f16*)(ws + 0 * MB), 128);
    hzero_k<<<(96 * 4 * PW + 255) / 256, 256, 0, stream>>>(ap2, 96);
    padx_k<<<(int)(((long)CIN * PLANE + 255) / 256), 256, 0, stream>>>(x, xpad);
    pack_a_k<<<(6 * 14 * 256 + 255) / 256, 256, 0, stream>>>(w2, pw2, 96, 432, 432, 1, 14);
    pack_a_k<<<(8 * 27 * 256 + 255) / 256, 256, 0, stream>>>(w3, pw3, 128, 864, 864, 1, 27);
    pack_a_k<<<(16 * 36 * 256 + 255) / 256, 256, 0, stream>>>(fw1, pf1, 256, 1152, 1, 256, 36);
    pack_a_k<<<(8 * 8 * 256 + 255) / 256, 256, 0, stream>>>(fw2, pf2, 128, 256, 1, 128, 8);

    // stage 1: conv1 direct -> ap1 (padded), GN in place
    conv1_k<<<dim3(HWPIX / 256, 48), 256, 0, stream>>>(xpad, w1, b1, ap1);
    gstats_k<<<6 * BPG, 256, 0, stream>>>(ap1, part, 8);
    gfinal_k<<<1, 64, 0, stream>>>(part, stats, 6);
    gn_apply_k<<<48 * HWPIX / 256, 256, 0, stream>>>(ap1, stats, g1, be1, 8);

    // stage 2: conv2 WMMA (6 waves/block) -> ap2, GN in place
    gemm_k<0><<<HWPIX / 64, 192, 0, stream>>>(
        (const v4u*)pw2, ap1, b2, ap2, 96, 14, nullptr, nullptr, 0, 0);
    gstats_k<<<12 * BPG, 256, 0, stream>>>(ap2, part, 8);
    gfinal_k<<<1, 64, 0, stream>>>(part, stats, 12);
    gn_apply_k<<<96 * HWPIX / 256, 256, 0, stream>>>(ap2, stats, g2, be2, 8);

    // stage 3: conv3 WMMA (8 waves/block) -> ap3 (reuses R0; ap1 dead), GN in place
    gemm_k<0><<<HWPIX / 64, 256, 0, stream>>>(
        (const v4u*)pw3, ap2, b3, ap3, 128, 27, nullptr, nullptr, 0, 0);
    gstats_k<<<16 * BPG, 256, 0, stream>>>(ap3, part, 8);
    gfinal_k<<<1, 64, 0, stream>>>(part, stats, 16);
    gn_apply_k<<<128 * HWPIX / 256, 256, 0, stream>>>(ap3, stats, g3, be3, 8);

    // value head
    gstats_k<<<128 * BPG, 256, 0, stream>>>(ap3, part, 1);
    gfinal_k<<<1, 128, 0, stream>>>(part, stats, 128);
    value_k<<<1, 64, 0, stream>>>(stats, vw1, vb1, vw2, vb2, out + 262146);

    // per-cell MLP
    gemm_k<1><<<NCELLS / 64, 512, 0, stream>>>(
        (const v4u*)pf1, ap3, fb1, hid1, 256, 36, cell_i, cell_j, 0, 1);
    gemm_k<2><<<NCELLS / 64, 256, 0, stream>>>(
        (const v4u*)pf2, hid1, fb2, shbuf, 128, 8, nullptr, nullptr, 256, 1);

    // heads (TDM-staged weights when available) + deterministic reductions + scatter
    heads_k<<<NCELLS / 256, 256, 0, stream>>>(shbuf, cell_i, cell_j, action,
                                              bw, bb, iw, ib, tw, tb, out, hpart);
    lpent_final_k<<<1, 32, 0, stream>>>(hpart, out, NCELLS / 256);
}
